// GraphNeuralNetworkEnsemble_25211458028038
// MI455X (gfx1250) — matrix operand order repaired and checked by reading, not compiled, not run on hardware
//
#include <hip/hip_runtime.h>
#include <hip/hip_bf16.h>

typedef __attribute__((ext_vector_type(16))) _Float16 v16h;
typedef __attribute__((ext_vector_type(8)))  _Float16 v8h;
typedef __attribute__((ext_vector_type(8)))  float    v8f;

#define LN_EPS 1e-5f
#define NB 128          // batch
#define NN 32           // nodes
#define GOUT_SZ (5*128*32)

// ---- WMMA fragment helpers (CDNA5 16x16x32 f16 -> f32) -------------------
// A 16x32 f16: lane L holds row M=L&15; halves v: K = (v&7) + ((v>>3)<<4) + ((L>>4)<<3)
// => two contiguous 16-byte runs per lane: [kl .. kl+7] and [kl+16 .. kl+23]
__device__ __forceinline__ v16h ldA(const _Float16* A, int lda, int m0, int k0, int lane) {
  const int m  = m0 + (lane & 15);
  const int kl = (lane >> 4) << 3;
  const _Float16* p = A + m * lda + k0 + kl;
  const v8h lo = *(const v8h*)(p);        // K = kl+0..7   (16B aligned)
  const v8h hi = *(const v8h*)(p + 16);   // K = kl+16..23 (16B aligned)
  return __builtin_shufflevector(lo, hi, 0,1,2,3,4,5,6,7, 8,9,10,11,12,13,14,15);
}
// B 32x16 (KxN) from global f32 (weights, L2-resident), cvt to f16 in regs.
// lane L holds col N=L&15; halves v: K = k0 + (v&7) + ((v>>3)<<4) + ((L>>4)<<3)
__device__ __forceinline__ v16h ldB(const float* B, int ldb, int k0, int n0, int kmax, int lane) {
  const int nn = n0 + (lane & 15);
  const int kb = k0 + ((lane >> 4) << 3);
  v16h b;
#pragma unroll
  for (int v = 0; v < 16; ++v) {
    const int k = kb + (v & 7) + ((v >> 3) << 4);
    b[v] = (k < kmax) ? (_Float16)B[k * ldb + nn] : (_Float16)0.0f;
  }
  return b;
}
// C/D 16x16 f32: VGPR r, lane L -> row = r + ((L>>4)<<3), col = L&15
__device__ __forceinline__ void stC(float* C, int ldc, int m0, int n0, v8f c, int lane) {
  const int nn = n0 + (lane & 15);
  const int rb = m0 + ((lane >> 4) << 3);
#pragma unroll
  for (int r = 0; r < 8; ++r) C[(rb + r) * ldc + nn] = c[r];
}

__device__ __forceinline__ v8f wmma_f16(v16h a, v16h b, v8f c) {
  return __builtin_amdgcn_wmma_f32_16x16x32_f16(false, a, false, b, (short)0, c, false, false);
}

__global__ __launch_bounds__(256)
void gnn_ens_kernel(const float* __restrict__ agent, const float* __restrict__ dyn,
                    const float* __restrict__ stat,  const float* __restrict__ act,
                    const float* __restrict__ eW1, const float* __restrict__ eb1,
                    const float* __restrict__ eg1, const float* __restrict__ ebe1,
                    const float* __restrict__ eW2, const float* __restrict__ eb2,
                    const float* __restrict__ nW1, const float* __restrict__ nb1,
                    const float* __restrict__ ng1, const float* __restrict__ nbe1,
                    const float* __restrict__ nW2, const float* __restrict__ nb2,
                    const float* __restrict__ gW1, const float* __restrict__ gb1,
                    const float* __restrict__ gg1, const float* __restrict__ gbe1,
                    const float* __restrict__ gW2, const float* __restrict__ gb2,
                    float* __restrict__ out) {
  // fp32 activation buffers
  __shared__ float sP[NN * 128];      // attr_i @ eW1[0:48]
  __shared__ float sQ[NN * 128];      // attr_j @ eW1[48:96]
  __shared__ float sSrow[NN * 128];   // sum_{j!=i} relu(ln(h_ij))
  __shared__ float sAgg[NN * 128];    // agg @ eW2 ; reused as node-hidden
  __shared__ float sR[128];
  __shared__ float sMu[1024], sRstd[1024];
  __shared__ float sGagg[128];
  __shared__ float sAg[32], sAc[16];
  __shared__ float sGh[128], sGact[128];
  __shared__ float sSallP[256];
  __shared__ float sB1[128], sG1[128], sBe1[128];
  // f16 GEMM operands
  __shared__ _Float16 sAttrA[NN * 64];    // attr padded K->64
  __shared__ _Float16 sAggA[NN * 128];    // agg_hidden f16 ; reused as relu'd node hidden
  __shared__ _Float16 sNodeIn[NN * 224];  // [attr(48)|agent,act(48)|agg(128)]

  const int ens = blockIdx.x >> 7;
  const int b   = blockIdx.x & 127;
  const int tid = threadIdx.x;
  const int lane = tid & 31;
  // Scalarize the wave id: makes every tile loop wave-uniform in SGPRs, so the
  // compiler emits scalar branches (no EXEC predication around v_wmma).
  const int wave = __builtin_amdgcn_readfirstlane(tid >> 5);

  const float* Ag  = agent + (ens * NB + b) * 32;
  const float* Ac  = act   + (ens * NB + b) * 16;
  const float* Dy  = dyn   + ((ens * NB + b) * NN) * 32;
  const float* St  = stat  + ((ens * NB + b) * NN) * 16;
  const float* EW1 = eW1 + ens * 144 * 128;
  const float* EW2 = eW2 + ens * 128 * 128;
  const float* NW1 = nW1 + ens * 224 * 128;
  const float* NW2 = nW2 + ens * 128 * 32;

  // ---- P0: stage node attrs (f16) + agent/act -----------------------------
  for (int x = tid; x < NN * 64; x += 256) {
    const int i = x >> 6, k = x & 63;
    float v = 0.f;
    if (k < 32)      v = Dy[i * 32 + k];
    else if (k < 48) v = St[i * 16 + (k - 32)];
    sAttrA[x] = (_Float16)v;
    if (k < 48) sNodeIn[i * 224 + k] = (_Float16)v;
  }
  if (tid < 48) { float v = (tid < 32) ? Ag[tid] : Ac[tid - 32];
                  if (tid < 32) sAg[tid] = v; else sAc[tid - 32] = v; }
  __syncthreads();
  for (int x = tid; x < NN * 48; x += 256) {
    const int i = x / 48, k = x - i * 48;
    sNodeIn[i * 224 + 48 + k] = (_Float16)((k < 32) ? sAg[k] : sAc[k - 32]);
  }
  if (tid < 128) {  // R = (agent,act) @ eW1[96:144] + eb1
    float r = eb1[ens * 128 + tid];
    for (int k = 0; k < 32; ++k) r += sAg[k] * EW1[(96 + k) * 128 + tid];
    for (int k = 0; k < 16; ++k) r += sAc[k] * EW1[(128 + k) * 128 + tid];
    sR[tid] = r;
  }
  __syncthreads();

  // ---- P1: GEMM  [P|Q] = attr(32x48) @ [eW1[0:48] | eW1[48:96]] ----------
  for (int t = wave; t < 32; t += 8) {
    const int tm = t & 1, tn = t >> 1;
    const float* Bp = EW1 + ((tn < 8) ? 0 : 48 * 128);
    float* Cp = (tn < 8) ? sP : sQ;
    const int n0 = (tn & 7) * 16;
    v8f c = {};
#pragma unroll
    for (int k0 = 0; k0 < 64; k0 += 32)
      c = wmma_f16(ldA(sAttrA, 64, tm * 16, k0, lane),
                   ldB(Bp, 128, k0, n0, 48, lane), c);
    stC(Cp, 128, tm * 16, n0, c, lane);
  }
  __syncthreads();

  // ---- P2: per-edge LayerNorm stats (h_ij = P_i + Q_j + R) ----------------
#pragma unroll
  for (int cc = 0; cc < 4; ++cc) {
    const int e2 = tid + 256 * cc;
    const int i = e2 >> 5, j = e2 & 31;
    float s = 0.f, s2 = 0.f;
    for (int h = 0; h < 128; ++h) {
      const float v = sP[i * 128 + h] + sQ[j * 128 + h] + sR[h];
      s += v; s2 += v * v;
    }
    const float mu = s * (1.f / 128.f);
    sMu[e2] = mu;
    sRstd[e2] = rsqrtf(s2 * (1.f / 128.f) - mu * mu + LN_EPS);
  }
  __syncthreads();

  // ---- P3: relu(ln(h)) aggregated over j!=i (row mean) and all edges ------
  {
    const int h = tid & 127, ig = tid >> 7;
    const float g1h  = eg1[ens * 128 + h];
    const float be1h = ebe1[ens * 128 + h];
    const float Rh = sR[h];
    float sall = 0.f;
    for (int i = ig * 16; i < ig * 16 + 16; ++i) {
      const float Pih = sP[i * 128 + h];
      float acc = 0.f;
      for (int j = 0; j < 32; ++j) {
        if (j == i) continue;
        const int e2 = i * 32 + j;
        float v = Pih + sQ[j * 128 + h] + Rh;
        v = (v - sMu[e2]) * sRstd[e2] * g1h + be1h;
        acc += fmaxf(v, 0.f);
      }
      sSrow[i * 128 + h] = acc;
      sall += acc;
    }
    sSallP[ig * 128 + h] = sall;
  }
  __syncthreads();
  if (tid < 128) sGagg[tid] = (sSallP[tid] + sSallP[128 + tid]) * (1.f / 992.f);
  for (int x = tid; x < NN * 128; x += 256) sAggA[x] = (_Float16)(sSrow[x] * (1.f / 31.f));
  __syncthreads();

  // ---- P4: GEMM  agg = agg_hidden(32x128) @ eW2(128x128) ------------------
  for (int t = wave; t < 16; t += 8) {
    const int tm = t & 1, tn = t >> 1;
    v8f c = {};
#pragma unroll
    for (int k0 = 0; k0 < 128; k0 += 32)
      c = wmma_f16(ldA(sAggA, 128, tm * 16, k0, lane),
                   ldB(EW2, 128, k0, tn * 16, 128, lane), c);
    stC(sAgg, 128, tm * 16, tn * 16, c, lane);
  }
  __syncthreads();
  for (int x = tid; x < NN * 128; x += 256) {
    const int i = x >> 7, h = x & 127;
    sNodeIn[i * 224 + 96 + h] = (_Float16)(sAgg[x] + eb2[ens * 128 + h]);
  }
  if (tid < 128) { sB1[tid] = nb1[ens*128+tid]; sG1[tid] = ng1[ens*128+tid]; sBe1[tid] = nbe1[ens*128+tid]; }
  __syncthreads();

  // ---- P5: global MLP (M=1 -> VALU) ---------------------------------------
  if (tid < 128) {
    const float* GW1 = gW1 + ens * 176 * 128;
    float g = gb1[ens * 128 + tid];
    for (int k = 0; k < 32;  ++k) g += sAg[k]   * GW1[k * 128 + tid];
    for (int k = 0; k < 16;  ++k) g += sAc[k]   * GW1[(32 + k) * 128 + tid];
    for (int k = 0; k < 128; ++k) g += sGagg[k] * GW1[(48 + k) * 128 + tid];
    sGh[tid] = g;
  }
  __syncthreads();
  if (tid < 128) {
    float s = 0.f, s2 = 0.f;
    for (int k = 0; k < 128; ++k) { const float v = sGh[k]; s += v; s2 += v * v; }
    const float mu = s * (1.f / 128.f);
    const float rs = rsqrtf(s2 * (1.f / 128.f) - mu * mu + LN_EPS);
    const float v = (sGh[tid] - mu) * rs * gg1[ens * 128 + tid] + gbe1[ens * 128 + tid];
    sGact[tid] = fmaxf(v, 0.f);
  }
  __syncthreads();
  if (tid < 32) {
    const float* GW2 = gW2 + ens * 128 * 32;
    float o = gb2[ens * 32 + tid];
    for (int h = 0; h < 128; ++h) o += sGact[h] * GW2[h * 32 + tid];
    out[(ens * NB + b) * 32 + tid] = o;
  }

  // ---- P6: GEMM  hid = nodeIn(32x224) @ nW1(224x128); LN+ReLU -------------
  for (int t = wave; t < 16; t += 8) {
    const int tm = t & 1, tn = t >> 1;
    v8f c = {};
#pragma unroll
    for (int k0 = 0; k0 < 224; k0 += 32)
      c = wmma_f16(ldA(sNodeIn, 224, tm * 16, k0, lane),
                   ldB(NW1, 128, k0, tn * 16, 224, lane), c);
    stC(sAgg, 128, tm * 16, tn * 16, c, lane);
  }
  __syncthreads();
  if (tid < 32) {
    const int i = tid;
    float s = 0.f, s2 = 0.f;
    for (int h = 0; h < 128; ++h) {
      const float v = sAgg[i * 128 + h] + sB1[h];
      s += v; s2 += v * v;
    }
    const float mu = s * (1.f / 128.f);
    const float rs = rsqrtf(s2 * (1.f / 128.f) - mu * mu + LN_EPS);
    for (int h = 0; h < 128; ++h) {
      const float v = (sAgg[i * 128 + h] + sB1[h] - mu) * rs * sG1[h] + sBe1[h];
      sAggA[i * 128 + h] = (_Float16)fmaxf(v, 0.f);
    }
  }
  __syncthreads();

  // ---- P7: GEMM  node_out = hid(32x128) @ nW2(128x32) + nb2 ---------------
  float* nodeOut = out + GOUT_SZ + ((ens * NB + b) * NN) * 32;
  for (int t = wave; t < 4; t += 8) {
    const int tm = t & 1, tn = t >> 1;
    v8f c = {};
#pragma unroll
    for (int k0 = 0; k0 < 128; k0 += 32)
      c = wmma_f16(ldA(sAggA, 128, tm * 16, k0, lane),
                   ldB(NW2, 32, k0, tn * 16, 128, lane), c);
    const int nn = tn * 16 + (lane & 15);
    const int rb = tm * 16 + ((lane >> 4) << 3);
    const float bias = nb2[ens * 32 + nn];
#pragma unroll
    for (int r = 0; r < 8; ++r) nodeOut[(rb + r) * 32 + nn] = c[r] + bias;
  }
}

extern "C" void kernel_launch(void* const* d_in, const int* in_sizes, int n_in,
                              void* d_out, int out_size, void* d_ws, size_t ws_size,
                              hipStream_t stream) {
  (void)in_sizes; (void)n_in; (void)out_size; (void)d_ws; (void)ws_size;
  const float* agent = (const float*)d_in[0];
  const float* dyn   = (const float*)d_in[1];
  const float* stat  = (const float*)d_in[2];
  const float* act   = (const float*)d_in[3];
  const float* eW1 = (const float*)d_in[4];
  const float* eb1 = (const float*)d_in[5];
  const float* eg1 = (const float*)d_in[6];
  const float* ebe1= (const float*)d_in[7];
  const float* eW2 = (const float*)d_in[8];
  const float* eb2 = (const float*)d_in[9];
  const float* nW1 = (const float*)d_in[10];
  const float* nb1 = (const float*)d_in[11];
  const float* ng1 = (const float*)d_in[12];
  const float* nbe1= (const float*)d_in[13];
  const float* nW2 = (const float*)d_in[14];
  const float* nb2 = (const float*)d_in[15];
  const float* gW1 = (const float*)d_in[16];
  const float* gb1 = (const float*)d_in[17];
  const float* gg1 = (const float*)d_in[18];
  const float* gbe1= (const float*)d_in[19];
  const float* gW2 = (const float*)d_in[20];
  const float* gb2 = (const float*)d_in[21];

  gnn_ens_kernel<<<5 * 128, 256, 0, stream>>>(
      agent, dyn, stat, act,
      eW1, eb1, eg1, ebe1, eW2, eb2,
      nW1, nb1, ng1, nbe1, nW2, nb2,
      gW1, gb1, gg1, gbe1, gW2, gb2,
      (float*)d_out);
}
